// SupervisedHeteroSAGEModel_28896539968211
// MI455X (gfx1250) — matrix operand order repaired and checked by reading, not compiled, not run on hardware
//
#include <hip/hip_runtime.h>
#include <hip/hip_bf16.h>

// ---------------------------------------------------------------------------
// Types for CDNA5 WMMA (wave32): D(16x16 f32) = A(16x32 bf16) x B(32x16 bf16) + C
// ---------------------------------------------------------------------------
typedef __bf16  v16bf __attribute__((ext_vector_type(16)));
typedef float   v8f   __attribute__((ext_vector_type(8)));
typedef unsigned int v4u  __attribute__((ext_vector_type(4)));
typedef int          v8i  __attribute__((ext_vector_type(8)));
typedef int          v4i  __attribute__((ext_vector_type(4)));

#define HD 128   // hidden dim
#define OD 32    // output dim
#define FMD 64   // material feature dim

__device__ __forceinline__ v8f wmma_bf16(v16bf a, v16bf b, v8f c) {
  // 8 args: (neg_a, A, neg_b, B, c_mod, C, reuse_a, reuse_b)
  return __builtin_amdgcn_wmma_f32_16x16x32_bf16(false, a, false, b, (short)0, c,
                                                 false, false);
}

// Build an A-matrix fragment (16x32 bf16, contiguous k order) for this lane from a
// row of f32 data, scaling on conversion.
//   lanes 0-15 : row M=lane,    VGPR0..7 = K{kb..kb+15}, kb=K0
//   lanes16-31 : row M=lane-16, same with kb=K0+16
// Caller passes kb already including the +16 for the upper half-wave.
__device__ __forceinline__ v16bf load_a_frag(const float* __restrict__ row,
                                             int kb, float scale) {
  const float4* p = (const float4*)(row + kb);
  const float4* q = (const float4*)(row + kb + 8);
  float4 a0 = p[0], a1 = p[1];
  float4 b0 = q[0], b1 = q[1];
  float f[16] = {a0.x, a0.y, a0.z, a0.w, a1.x, a1.y, a1.z, a1.w,
                 b0.x, b0.y, b0.z, b0.w, b1.x, b1.y, b1.z, b1.w};
  v16bf v;
#pragma unroll
  for (int i = 0; i < 16; ++i) v[i] = (__bf16)(f[i] * scale);
  return v;
}

// ---------------------------------------------------------------------------
// Tensor Data Mover: DMA a 2D f32 tile (ncols x nrows, row-major, contiguous)
// from global memory into LDS.  D# per CDNA5 ISA.
// ---------------------------------------------------------------------------
#if __has_builtin(__builtin_amdgcn_tensor_load_to_lds)
__device__ __forceinline__ void tdm_load_2d_f32(unsigned lds_addr,
                                                const void* gptr,
                                                int ncols, int nrows) {
  unsigned long long ga = (unsigned long long)gptr;
  v4u g0;
  g0.x = 1u;                                           // count=1, user descriptor
  g0.y = lds_addr;                                     // LDS byte address
  g0.z = (unsigned)ga;                                 // global_addr[31:0]
  g0.w = (unsigned)((ga >> 32) & 0x1FFFFFFull) | (2u << 30);  // addr[56:32] | type=2
  v8i g1;
  g1[0] = (2 << 16);                                   // data_size=4B; no multicast
  g1[1] = (ncols & 0xFFFF) << 16;                      // tensor_dim0[15:0]
  g1[2] = ((ncols >> 16) & 0xFFFF) | ((nrows & 0xFFFF) << 16);  // dim0 hi | dim1 lo
  g1[3] = ((nrows >> 16) & 0xFFFF) | ((ncols & 0xFFFF) << 16);  // dim1 hi | tile_dim0
  g1[4] = (nrows & 0xFFFF);                            // tile_dim1 (tile_dim2=0)
  g1[5] = ncols;                                       // tensor_dim0_stride lo
  g1[6] = 0;
  g1[7] = 0;
  v4i z4 = {0, 0, 0, 0};
#if defined(__clang_major__) && (__clang_major__ >= 23)
  v8i z8 = {0, 0, 0, 0, 0, 0, 0, 0};
  __builtin_amdgcn_tensor_load_to_lds(g0, g1, z4, z4, z8, 0);
#else
  __builtin_amdgcn_tensor_load_to_lds(g0, g1, z4, z4, 0);
#endif
}
#endif

// ---------------------------------------------------------------------------
// Small utility kernels
// ---------------------------------------------------------------------------
__global__ void fill_zero_kernel(float* __restrict__ p, long long n) {
  long long i = (long long)blockIdx.x * blockDim.x + threadIdx.x;
  if (i < n) p[i] = 0.0f;
}

__global__ void degree_kernel(const int* __restrict__ dst,
                              float* __restrict__ deg, int E) {
  int e = blockIdx.x * blockDim.x + threadIdx.x;
  if (e < E) atomicAdd(&deg[dst[e]], 1.0f);
}

// x_e = emb_element[element_node_id]  (row gather, float4 vectorized)
__global__ void gather_rows_kernel(const float* __restrict__ src,
                                   const int* __restrict__ ids,
                                   float* __restrict__ dst, int N) {
  long long t = (long long)blockIdx.x * blockDim.x + threadIdx.x;
  if (t >= (long long)N * (HD / 4)) return;
  int row = (int)(t >> 5);
  int c   = (int)(t & 31);
  int id = ids[row];
  ((float4*)(dst + (size_t)row * HD))[c] =
      ((const float4*)(src + (size_t)id * HD))[c];
}

// agg[dst[e], :] += xsrc[src[e], :]   (segment-sum via f32 atomics; accumulator
// arrays fit in the 192MB L2 so atomic throughput is L2-bound, not HBM-bound)
__global__ void scatter_add_kernel(const float* __restrict__ xsrc,
                                   const int* __restrict__ src,
                                   const int* __restrict__ dst,
                                   float* __restrict__ agg, int E) {
  long long t = (long long)blockIdx.x * blockDim.x + threadIdx.x;
  if (t >= (long long)E * (HD / 4)) return;
  int e = (int)(t >> 5);
  int c = (int)(t & 31);
  int s = src[e];
  int d = dst[e];
  float4 v = ((const float4*)(xsrc + (size_t)s * HD))[c];
  float* a = agg + (size_t)d * HD + (size_t)c * 4;
  atomicAdd(a + 0, v.x);
  atomicAdd(a + 1, v.y);
  atomicAdd(a + 2, v.z);
  atomicAdd(a + 3, v.w);
}

// ---------------------------------------------------------------------------
// x_m = material_x @ lin_mat_W + b + emb_material[ids]      (K=64 WMMA GEMM)
// one wave -> 16x128 tile; 8 waves/WG -> 128 rows per block
// ---------------------------------------------------------------------------
__global__ __launch_bounds__(256) void encode_material_kernel(
    const float* __restrict__ X,     // [N,64]
    const float* __restrict__ W,     // [64,128]
    const float* __restrict__ bias,  // [128]
    const float* __restrict__ emb,   // [N,128]
    const int* __restrict__ ids,     // [N]
    float* __restrict__ out, int N) {
  __shared__ __bf16 Bm[HD * FMD];   // B as [n][k] bf16 -> contiguous lane frags
  __shared__ float bsh[HD];
  int tid = threadIdx.x, lane = tid & 31, wave = tid >> 5;

  for (int idx = tid; idx < FMD * HD; idx += 256) {
    int k = idx >> 7, n = idx & 127;      // W is [k][n]
    Bm[n * FMD + k] = (__bf16)W[idx];
  }
  if (tid < HD) bsh[tid] = bias[tid];
  __syncthreads();

  int rowBase = blockIdx.x * 128 + wave * 16;
  int m = rowBase + (lane & 15);
  int mc = m < N ? m : N - 1;
  const float* xrow = X + (size_t)mc * FMD;

  v8f acc[8] = {};
#pragma unroll
  for (int K0 = 0; K0 < FMD; K0 += 32) {
    int kb = K0 + ((lane >> 4) ? 16 : 0);
    v16bf a = load_a_frag(xrow, kb, 1.0f);
    int klds = K0 + ((lane >> 4) ? 16 : 0);
#pragma unroll
    for (int t = 0; t < 8; ++t) {
      v16bf b = *(const v16bf*)(Bm + (t * 16 + (lane & 15)) * FMD + klds);
      acc[t] = wmma_bf16(a, b, acc[t]);
    }
  }

  int rowTop = rowBase + ((lane >> 4) ? 8 : 0);
  int col = lane & 15;
#pragma unroll
  for (int r = 0; r < 8; ++r) {
    int row = rowTop + r;
    if (row < N) {
      int id = ids[row];
#pragma unroll
      for (int t = 0; t < 8; ++t) {
        int c = t * 16 + col;
        out[(size_t)row * HD + c] = acc[t][r] + bsh[c] + emb[(size_t)id * HD + c];
      }
    }
  }
}

// ---------------------------------------------------------------------------
// Fused SAGE conv:  out = relu( (agg/max(deg,1)) @ Wl + xdst @ Wr + b )
// = [mean | xdst] (N x 256) @ [Wl;Wr] (256 x 128), K=256 WMMA GEMM.
// Weights DMA'd to LDS f32 staging by the Tensor Data Mover, then converted
// to bf16 [n][k] layout.  Dynamic LDS: 64KB stage + 64KB bf16 B + bias.
// ---------------------------------------------------------------------------
__global__ __launch_bounds__(256) void sage_gemm_relu_kernel(
    const float* __restrict__ agg,   // [N,128]
    const float* __restrict__ deg,   // [N]
    const float* __restrict__ xdst,  // [N,128]
    const float* __restrict__ Wl,    // [128,128]
    const float* __restrict__ Wr,    // [128,128]
    const float* __restrict__ bias,  // [128]
    float* __restrict__ out, int N) {
  extern __shared__ char smem[];
  float*  stage = (float*)smem;               // 16384 f32 = 64 KB
  __bf16* Bm    = (__bf16*)(smem + 65536);    // [128][256] bf16 = 64 KB
  float*  bsh   = (float*)(smem + 131072);    // 128 f32
  int tid = threadIdx.x, lane = tid & 31, wave = tid >> 5;

  if (tid < HD) bsh[tid] = bias[tid];

#pragma unroll
  for (int phase = 0; phase < 2; ++phase) {
    const float* Wsrc = phase ? Wr : Wl;
#if __has_builtin(__builtin_amdgcn_tensor_load_to_lds)
    if (wave == 0) {
      tdm_load_2d_f32((unsigned)(unsigned long long)stage, Wsrc, HD, HD);
      __builtin_amdgcn_s_wait_tensorcnt(0);
    }
#else
    for (int idx = tid; idx < HD * HD; idx += 256) stage[idx] = Wsrc[idx];
#endif
    __syncthreads();
    for (int idx = tid; idx < HD * HD; idx += 256) {
      int k = idx >> 7, n = idx & 127;       // stage is [k][n]
      Bm[n * 256 + (phase ? HD : 0) + k] = (__bf16)stage[idx];
    }
    __syncthreads();
  }

  int rowBase = blockIdx.x * 128 + wave * 16;
  int m = rowBase + (lane & 15);
  int mc = m < N ? m : N - 1;
  const float* arow = agg + (size_t)mc * HD;
  const float* xrow = xdst + (size_t)mc * HD;
  float rdeg = 1.0f / fmaxf(deg[mc], 1.0f);

  v8f acc[8] = {};
#pragma unroll
  for (int K0 = 0; K0 < 2 * HD; K0 += 32) {
    int kb = (K0 & (HD - 1)) + ((lane >> 4) ? 16 : 0);
    v16bf a = (K0 < HD) ? load_a_frag(arow, kb, rdeg)   // mean part
                        : load_a_frag(xrow, kb, 1.0f);  // self part
    int klds = K0 + ((lane >> 4) ? 16 : 0);
#pragma unroll
    for (int t = 0; t < 8; ++t) {
      v16bf b = *(const v16bf*)(Bm + (t * 16 + (lane & 15)) * 256 + klds);
      acc[t] = wmma_bf16(a, b, acc[t]);
    }
  }

  int rowTop = rowBase + ((lane >> 4) ? 8 : 0);
  int col = lane & 15;
#pragma unroll
  for (int r = 0; r < 8; ++r) {
    int row = rowTop + r;
    if (row < N) {
#pragma unroll
      for (int t = 0; t < 8; ++t) {
        int c = t * 16 + col;
        float v = acc[t][r] + bsh[c];
        out[(size_t)row * HD + c] = fmaxf(v, 0.0f);
      }
    }
  }
}

// ---------------------------------------------------------------------------
// Final projection: d_out = g_m @ out_W + out_b   ([N,128]@[128,32])
// ---------------------------------------------------------------------------
__global__ __launch_bounds__(256) void out_gemm_kernel(
    const float* __restrict__ X,     // [N,128]
    const float* __restrict__ W,     // [128,32]
    const float* __restrict__ bias,  // [32]
    float* __restrict__ out, int N) {
  __shared__ __bf16 Bm[OD * HD];    // [n][k]
  __shared__ float bsh[OD];
  int tid = threadIdx.x, lane = tid & 31, wave = tid >> 5;

  for (int idx = tid; idx < HD * OD; idx += 256) {
    int k = idx >> 5, n = idx & 31;       // W is [k][n]
    Bm[n * HD + k] = (__bf16)W[idx];
  }
  if (tid < OD) bsh[tid] = bias[tid];
  __syncthreads();

  int rowBase = blockIdx.x * 128 + wave * 16;
  int m = rowBase + (lane & 15);
  int mc = m < N ? m : N - 1;
  const float* xrow = X + (size_t)mc * HD;

  v8f acc[2] = {};
#pragma unroll
  for (int K0 = 0; K0 < HD; K0 += 32) {
    int kb = K0 + ((lane >> 4) ? 16 : 0);
    v16bf a = load_a_frag(xrow, kb, 1.0f);
    int klds = K0 + ((lane >> 4) ? 16 : 0);
#pragma unroll
    for (int t = 0; t < 2; ++t) {
      v16bf b = *(const v16bf*)(Bm + (t * 16 + (lane & 15)) * HD + klds);
      acc[t] = wmma_bf16(a, b, acc[t]);
    }
  }

  int rowTop = rowBase + ((lane >> 4) ? 8 : 0);
  int col = lane & 15;
#pragma unroll
  for (int r = 0; r < 8; ++r) {
    int row = rowTop + r;
    if (row < N) {
#pragma unroll
      for (int t = 0; t < 2; ++t) {
        int c = t * 16 + col;
        out[(size_t)row * OD + c] = acc[t][r] + bsh[c];
      }
    }
  }
}

// ---------------------------------------------------------------------------
// Host-side orchestration
// ---------------------------------------------------------------------------
extern "C" void kernel_launch(void* const* d_in, const int* in_sizes, int n_in,
                              void* d_out, int out_size, void* d_ws, size_t ws_size,
                              hipStream_t stream) {
  const int NM = in_sizes[1];      // 100000
  const int NE = in_sizes[2];      // 100000
  const int E1 = in_sizes[3] / 2;  // m2e edges
  const int E2 = in_sizes[4] / 2;  // e2m edges

  const float* material_x = (const float*)d_in[0];
  const int*   mat_id     = (const int*)d_in[1];
  const int*   ele_id     = (const int*)d_in[2];
  const int*   ei_m2e     = (const int*)d_in[3];  // [src_m | dst_e]
  const int*   ei_e2m     = (const int*)d_in[4];  // [src_e | dst_m]
  const float* emb_m      = (const float*)d_in[5];
  const float* emb_e      = (const float*)d_in[6];
  const float* lin_W      = (const float*)d_in[7];
  const float* lin_b      = (const float*)d_in[8];
  const float* out_W      = (const float*)d_in[9];
  const float* out_b      = (const float*)d_in[10];
  const float* l1_m2e_Wl = (const float*)d_in[11];
  const float* l1_m2e_Wr = (const float*)d_in[12];
  const float* l1_m2e_b  = (const float*)d_in[13];
  const float* l1_e2m_Wl = (const float*)d_in[14];
  const float* l1_e2m_Wr = (const float*)d_in[15];
  const float* l1_e2m_b  = (const float*)d_in[16];
  // l2_m2e (17..19) feeds only g_e, which is dead in the reference -> skipped
  const float* l2_e2m_Wl = (const float*)d_in[20];
  const float* l2_e2m_Wr = (const float*)d_in[21];
  const float* l2_e2m_b  = (const float*)d_in[22];

  // workspace carve-up (f32)
  float* ws = (float*)d_ws;
  float* x_m   = ws; ws += (size_t)NM * HD;
  float* x_e   = ws; ws += (size_t)NE * HD;
  float* agg_m = ws; ws += (size_t)NM * HD;
  float* agg_e = ws; ws += (size_t)NE * HD;
  float* h_m   = ws; ws += (size_t)NM * HD;
  float* h_e   = ws; ws += (size_t)NE * HD;
  float* deg_m = ws; ws += NM;
  float* deg_e = ws; ws += NE;
  float* g_m   = x_m;  // x_m dead after layer 1 -> reuse for layer-2 output

  auto cdiv = [](long long a, long long b) { return (int)((a + b - 1) / b); };
  const int rowBlocksM = cdiv(NM, 128);
  const int rowBlocksE = cdiv(NE, 128);
  const size_t sageSmem = 65536 + 65536 + 512;

  // ---- init accumulators / degrees ----
  fill_zero_kernel<<<cdiv(NM, 256), 256, 0, stream>>>(deg_m, NM);
  fill_zero_kernel<<<cdiv(NE, 256), 256, 0, stream>>>(deg_e, NE);
  fill_zero_kernel<<<cdiv((long long)NM * HD, 256), 256, 0, stream>>>(agg_m, (long long)NM * HD);
  fill_zero_kernel<<<cdiv((long long)NE * HD, 256), 256, 0, stream>>>(agg_e, (long long)NE * HD);
  degree_kernel<<<cdiv(E2, 256), 256, 0, stream>>>(ei_e2m + E2, deg_m, E2);
  degree_kernel<<<cdiv(E1, 256), 256, 0, stream>>>(ei_m2e + E1, deg_e, E1);

  // ---- node-type input encoders ----
  encode_material_kernel<<<rowBlocksM, 256, 0, stream>>>(material_x, lin_W, lin_b,
                                                         emb_m, mat_id, x_m, NM);
  gather_rows_kernel<<<cdiv((long long)NE * 32, 256), 256, 0, stream>>>(emb_e, ele_id, x_e, NE);

  // ---- layer 1: scatter then fused GEMMs ----
  scatter_add_kernel<<<cdiv((long long)E2 * 32, 256), 256, 0, stream>>>(
      x_e, ei_e2m, ei_e2m + E2, agg_m, E2);
  scatter_add_kernel<<<cdiv((long long)E1 * 32, 256), 256, 0, stream>>>(
      x_m, ei_m2e, ei_m2e + E1, agg_e, E1);
  sage_gemm_relu_kernel<<<rowBlocksM, 256, sageSmem, stream>>>(
      agg_m, deg_m, x_m, l1_e2m_Wl, l1_e2m_Wr, l1_e2m_b, h_m, NM);
  sage_gemm_relu_kernel<<<rowBlocksE, 256, sageSmem, stream>>>(
      agg_e, deg_e, x_e, l1_m2e_Wl, l1_m2e_Wr, l1_m2e_b, h_e, NE);

  // ---- layer 2 (material side only; element side is dead code) ----
  fill_zero_kernel<<<cdiv((long long)NM * HD, 256), 256, 0, stream>>>(agg_m, (long long)NM * HD);
  scatter_add_kernel<<<cdiv((long long)E2 * 32, 256), 256, 0, stream>>>(
      h_e, ei_e2m, ei_e2m + E2, agg_m, E2);
  sage_gemm_relu_kernel<<<rowBlocksM, 256, sageSmem, stream>>>(
      agg_m, deg_m, h_m, l2_e2m_Wl, l2_e2m_Wr, l2_e2m_b, g_m, NM);

  // ---- final projection ----
  out_gemm_kernel<<<rowBlocksM, 256, 0, stream>>>(g_m, out_W, out_b, (float*)d_out, NM);

  (void)n_in; (void)out_size; (void)ws_size;
}
